// SpectralTemporalNet_81217831568102
// MI455X (gfx1250) — matrix operand order repaired and checked by reading, not compiled, not run on hardware
//
#include <hip/hip_runtime.h>

typedef __attribute__((ext_vector_type(16))) _Float16 v16h;
typedef __attribute__((ext_vector_type(8)))  float    v8f;

#define DIMD   256
#define DFEAT  512
#define NHEAD  8
#define HDIM   64

__device__ __forceinline__ v8f wmma16(v16h a, v16h b, v8f c) {
  // D = A(16x32 f16) * B(32x16 f16) + C(16x16 f32)
  return __builtin_amdgcn_wmma_f32_16x16x32_f16(false, a, false, b, (short)0, c, false, false);
}

// ---------------------------------------------------------------- utilities
__global__ void zero_kernel(float* __restrict__ p, int n) {
  int i = blockIdx.x * blockDim.x + threadIdx.x;
  if (i < n) p[i] = 0.0f;
}

__global__ void copy_kernel(const float* __restrict__ s, float* __restrict__ d, int n) {
  int i = blockIdx.x * blockDim.x + threadIdx.x;
  if (i < n) d[i] = s[i];
}

__global__ void deg_kernel(const int* __restrict__ dst, float* __restrict__ deg, int E) {
  int e = blockIdx.x * blockDim.x + threadIdx.x;
  if (e < E) atomicAdd(&deg[dst[e]], 1.0f);
}

__global__ void dis_kernel(const float* __restrict__ deg, float* __restrict__ dis, int n) {
  int i = blockIdx.x * blockDim.x + threadIdx.x;
  if (i < n) {
    float d = deg[i];
    dis[i] = (d > 0.0f) ? rsqrtf(fmaxf(d, 1e-12f)) : 0.0f;
  }
}

// OUT[dst] += w * V[src], w = 1 or -(dis[src]*dis[dst])
__global__ void scatter_add_kernel(const float* __restrict__ V, const int* __restrict__ src,
                                   const int* __restrict__ dst, const float* __restrict__ dis,
                                   float* __restrict__ OUT, int E, int D) {
  int idx = blockIdx.x * blockDim.x + threadIdx.x;
  int perEdge = D >> 2;
  if (idx >= E * perEdge) return;
  int e = idx / perEdge;
  int f = (idx - e * perEdge) << 2;
  int s = src[e], d = dst[e];
  float w = 1.0f;
  if (dis) w = -(dis[s] * dis[d]);
  const float4 vv = *reinterpret_cast<const float4*>(V + (size_t)s * D + f);
  float* o = OUT + (size_t)d * D + f;
  atomicAdd(o + 0, w * vv.x);
  atomicAdd(o + 1, w * vv.y);
  atomicAdd(o + 2, w * vv.z);
  atomicAdd(o + 3, w * vv.w);
}

// Y = 2*Y - S  (Chebyshev recurrence)
__global__ void combine_kernel(float* __restrict__ Y, const float* __restrict__ S, int n) {
  int i = blockIdx.x * blockDim.x + threadIdx.x;
  if (i < n) Y[i] = 2.0f * Y[i] - S[i];
}

// BatchNorm apply (+optional relu), strided output write
__global__ void bn_apply_kernel(const float* __restrict__ Z, const float* __restrict__ csum,
                                const float* __restrict__ csumsq, const float* __restrict__ g,
                                const float* __restrict__ b, float* __restrict__ OUT,
                                int n, int D, int ldOut, int relu) {
  int idx = blockIdx.x * blockDim.x + threadIdx.x;
  if (idx >= n * D) return;
  int r = idx / D, c = idx - r * D;
  float inv_n = 1.0f / (float)n;
  float mu = csum[c] * inv_n;
  float var = csumsq[c] * inv_n - mu * mu;
  float v = g[c] * (Z[idx] - mu) * rsqrtf(var + 1e-5f) + b[c];
  if (relu) v = fmaxf(v, 0.0f);
  OUT[(size_t)r * ldOut + c] = v;
}

// Row LayerNorm: one wave per row of D=256
__global__ void layernorm_kernel(const float* __restrict__ X, const float* __restrict__ g,
                                 const float* __restrict__ b, float* __restrict__ OUT, int D) {
  int row = blockIdx.x;
  int lane = threadIdx.x;
  float vals[8];
  float sum = 0.0f;
#pragma unroll
  for (int j = 0; j < 8; ++j) {
    vals[j] = X[(size_t)row * D + j * 32 + lane];
    sum += vals[j];
  }
#pragma unroll
  for (int o = 16; o > 0; o >>= 1) sum += __shfl_xor(sum, o, 32);
  float mu = sum / (float)D;
  float vs = 0.0f;
#pragma unroll
  for (int j = 0; j < 8; ++j) { float dlt = vals[j] - mu; vs += dlt * dlt; }
#pragma unroll
  for (int o = 16; o > 0; o >>= 1) vs += __shfl_xor(vs, o, 32);
  float inv = rsqrtf(vs / (float)D + 1e-5f);
#pragma unroll
  for (int j = 0; j < 8; ++j) {
    int c = j * 32 + lane;
    OUT[(size_t)row * D + c] = g[c] * (vals[j] - mu) * inv + b[c];
  }
}

// ---------------------------------------------------------------- fragment packing
// A fragments: (M/16 x K/32) tiles, per tile 32 lanes x 16 halves in WMMA A-layout.
__global__ void pack_a_kernel(const float* __restrict__ A, _Float16* __restrict__ aP,
                              int M, int K, int ldA) {
  int idx = blockIdx.x * blockDim.x + threadIdx.x;
  int KC = K >> 5;
  if (idx >= (M >> 4) * KC * 32) return;
  int lane = idx & 31;
  int frag = idx >> 5;
  int tm = frag / KC;
  int kc = frag - tm * KC;
  int ar = tm * 16 + (lane & 15);
  int kh = (lane >> 4) << 4;
  v16h a;
#pragma unroll
  for (int v = 0; v < 8; ++v) {
    int kk = (kc << 5) + (v << 1) + kh;
    float2 p = *reinterpret_cast<const float2*>(A + (size_t)ar * ldA + kk);
    a[2 * v]     = (_Float16)p.x;
    a[2 * v + 1] = (_Float16)p.y;
  }
  *reinterpret_cast<v16h*>(aP + ((size_t)frag * 32 + lane) * 16) = a;
}

// B fragments: (K/32 x Nc/16) tiles in WMMA B-layout; frag index = kc*NT + tn.
__global__ void pack_b_kernel(const float* __restrict__ W, _Float16* __restrict__ bP,
                              int K, int Nc) {
  int idx = blockIdx.x * blockDim.x + threadIdx.x;
  int NT = Nc >> 4;
  if (idx >= (K >> 5) * NT * 32) return;
  int lane = idx & 31;
  int frag = idx >> 5;
  int kc = frag / NT;
  int tn = frag - kc * NT;
  int bn = tn * 16 + (lane & 15);
  int kgrp = (lane >> 4) << 4;
  v16h b;
#pragma unroll
  for (int v = 0; v < 8; ++v) {
    int kr = (kc << 5) + kgrp + (v << 1);
    b[2 * v]     = (_Float16)W[(size_t)kr * Nc + bn];
    b[2 * v + 1] = (_Float16)W[(size_t)(kr + 1) * Nc + bn];
  }
  *reinterpret_cast<v16h*>(bP + ((size_t)frag * 32 + lane) * 16) = b;
}

// K fragments for attention scores: frag = ((h*(n/16)+kt)*2 + c); B-layout 32(d) x 16(keys).
__global__ void pack_k_kernel(const float* __restrict__ qkv, _Float16* __restrict__ kP, int n) {
  int idx = blockIdx.x * blockDim.x + threadIdx.x;
  int nkt = n >> 4;
  if (idx >= NHEAD * nkt * 2 * 32) return;
  int lane = idx & 31;
  int frag = idx >> 5;
  int c = frag & 1;
  int t = frag >> 1;
  int kt = t % nkt;
  int h = t / nkt;
  int key = kt * 16 + (lane & 15);
  int kgrp = (lane >> 4) << 4;
  v16h b;
#pragma unroll
  for (int v = 0; v < 8; ++v) {
    int d = c * 32 + kgrp + (v << 1);
    float2 p = *reinterpret_cast<const float2*>(qkv + (size_t)key * (3 * DFEAT) + DFEAT + h * HDIM + d);
    b[2 * v]     = (_Float16)p.x;
    b[2 * v + 1] = (_Float16)p.y;
  }
  *reinterpret_cast<v16h*>(kP + ((size_t)frag * 32 + lane) * 16) = b;
}

// V fragments for P@V: frag = ((h*(n/32)+kc)*4 + q4); B-layout 32(keys) x 16(d).
__global__ void pack_v_kernel(const float* __restrict__ qkv, _Float16* __restrict__ vP, int n) {
  int idx = blockIdx.x * blockDim.x + threadIdx.x;
  int nkc = n >> 5;
  if (idx >= NHEAD * nkc * 4 * 32) return;
  int lane = idx & 31;
  int frag = idx >> 5;
  int q4 = frag & 3;
  int t = frag >> 2;
  int kc = t % nkc;
  int h = t / nkc;
  int d = q4 * 16 + (lane & 15);
  int kgrp = (lane >> 4) << 4;
  v16h b;
#pragma unroll
  for (int v = 0; v < 8; ++v) {
    int krow = kc * 32 + kgrp + (v << 1);
    const float* vp = qkv + (size_t)krow * (3 * DFEAT) + 2 * DFEAT + h * HDIM + d;
    b[2 * v]     = (_Float16)vp[0];
    b[2 * v + 1] = (_Float16)vp[3 * DFEAT];
  }
  *reinterpret_cast<v16h*>(vP + ((size_t)frag * 32 + lane) * 16) = b;
}

// ---------------------------------------------------------------- WMMA GEMM on packed fragments
// One wave computes a 16x64 C strip (4 wmma sharing one A fragment per K-chunk).
__global__ void wmma_gemm_pk(const _Float16* __restrict__ aP, const _Float16* __restrict__ bP,
                             const float* __restrict__ bias, float* __restrict__ OUT,
                             int M, int K, int Nc, int ldC, int relu, int accum,
                             float* __restrict__ colsum, float* __restrict__ colsumsq) {
  const int lane = threadIdx.x & 31;
  const int wid  = (blockIdx.x << 3) + (threadIdx.x >> 5);
  const int KC = K >> 5, NT = Nc >> 4, NG = Nc >> 6;
  if (wid >= (M >> 4) * NG) return;
  const int tm = wid / NG;
  const int tg = wid - tm * NG;
  const int rbase = (lane >> 4) << 3;

  v8f acc[4];
#pragma unroll
  for (int j = 0; j < 4; ++j)
#pragma unroll
    for (int r8 = 0; r8 < 8; ++r8) acc[j][r8] = 0.0f;
  if (accum) {
#pragma unroll
    for (int j = 0; j < 4; ++j)
#pragma unroll
      for (int r8 = 0; r8 < 8; ++r8)
        acc[j][r8] = OUT[(size_t)(tm * 16 + r8 + rbase) * ldC + tg * 64 + j * 16 + (lane & 15)];
  }

  const _Float16* ap = aP + ((size_t)tm * KC * 32 + lane) * 16;
  const _Float16* bp = bP + ((size_t)tg * 4 * 32 + lane) * 16;
  for (int kc = 0; kc < KC; ++kc) {
    if (kc + 1 < KC) {
      __builtin_prefetch(ap + (size_t)(kc + 1) * 512, 0, 1);
      __builtin_prefetch(bp + (size_t)(kc + 1) * NT * 512, 0, 1);
    }
    v16h a = *reinterpret_cast<const v16h*>(ap + (size_t)kc * 512);
#pragma unroll
    for (int j = 0; j < 4; ++j) {
      v16h b = *reinterpret_cast<const v16h*>(bp + ((size_t)kc * NT + j) * 512);
      acc[j] = wmma16(a, b, acc[j]);
    }
  }

#pragma unroll
  for (int j = 0; j < 4; ++j) {
    int bn = tg * 64 + j * 16 + (lane & 15);
    float bv = bias ? bias[bn] : 0.0f;
    float psum = 0.0f, psq = 0.0f;
#pragma unroll
    for (int r8 = 0; r8 < 8; ++r8) {
      float v = acc[j][r8] + bv;
      if (relu) v = fmaxf(v, 0.0f);
      acc[j][r8] = v;
      psum += v;
      psq += v * v;
    }
#pragma unroll
    for (int r8 = 0; r8 < 8; ++r8)
      OUT[(size_t)(tm * 16 + r8 + rbase) * ldC + bn] = acc[j][r8];
    if (colsum) {  // lanes l and l+16 share column bn
      psum += __shfl_xor(psum, 16, 32);
      psq  += __shfl_xor(psq, 16, 32);
      if (lane < 16) {
        atomicAdd(&colsum[bn], psum);
        atomicAdd(&colsumsq[bn], psq);
      }
    }
  }
}

// ---------------------------------------------------------------- Flash attention
// One wave per (head, 16-query tile); 32-key KV tiles double-buffered in LDS via
// global_load_async_to_lds_b128 (ASYNCcnt pipeline). 8 wmma per KV tile.
__global__ void flash_attn_kernel(const float* __restrict__ qkv,
                                  const _Float16* __restrict__ kP,
                                  const _Float16* __restrict__ vP,
                                  float* __restrict__ out, int n) {
  // NOTE: declared first so it sits at LDS offset 0 (async dst addresses assume this).
  __shared__ __align__(32) _Float16 kv_lds[2][8][512];  // 16 KB double buffer
  __shared__ __align__(32) _Float16 lds_p[16 * 32];     // P transpose staging

  const int qtiles = n >> 4;
  const int h  = blockIdx.x / qtiles;
  const int qt = blockIdx.x - h * qtiles;
  const int lane = threadIdx.x & 31;
  const int qrow0 = qt * 16;
  const int LD = 3 * DFEAT;
  const float scale = 0.125f;  // 1/sqrt(64)

  const int bn = lane & 15;
  const int kh = (lane >> 4) << 4;
  const int rbase = (lane >> 4) << 3;
  const int nkt = n >> 4, nkc = n >> 5;
  const size_t hK = (size_t)h * nkt, hV = (size_t)h * nkc;

  // Preload Q as two A-fragments (inner dim 64 = 2 chunks of 32), fold softmax scale.
  v16h aq[2];
  const int ar = qrow0 + bn;
#pragma unroll
  for (int c = 0; c < 2; ++c) {
#pragma unroll
    for (int v = 0; v < 8; ++v) {
      int kk = c * 32 + (v << 1) + kh;
      float2 p = *reinterpret_cast<const float2*>(qkv + (size_t)ar * LD + h * HDIM + kk);
      aq[c][2 * v]     = (_Float16)(p.x * scale);
      aq[c][2 * v + 1] = (_Float16)(p.y * scale);
    }
  }

  // Async-stage the 8 fragments (4 K + 4 V, 1KB each) of one 32-key tile into LDS.
  auto issue = [&](int kb, int buf) {
    int kt = kb >> 4, kc = kb >> 5;
    const _Float16* g[8];
    g[0] = kP + ((hK + kt) * 2 + 0) * 512;
    g[1] = kP + ((hK + kt) * 2 + 1) * 512;
    g[2] = kP + ((hK + kt + 1) * 2 + 0) * 512;
    g[3] = kP + ((hK + kt + 1) * 2 + 1) * 512;
    g[4] = vP + ((hV + kc) * 4 + 0) * 512;
    g[5] = vP + ((hV + kc) * 4 + 1) * 512;
    g[6] = vP + ((hV + kc) * 4 + 2) * 512;
    g[7] = vP + ((hV + kc) * 4 + 3) * 512;
#pragma unroll
    for (int f = 0; f < 8; ++f) {
      const char* gp = (const char*)g[f] + lane * 32;
      unsigned la = (unsigned)(buf * 8192 + f * 1024 + lane * 32);
      asm volatile("global_load_async_to_lds_b128 %0, %1, off" :: "v"(la), "v"(gp) : "memory");
      asm volatile("global_load_async_to_lds_b128 %0, %1, off" :: "v"(la + 16), "v"(gp + 16) : "memory");
    }
  };

  v8f m_run, l_run, accO[4];
#pragma unroll
  for (int r8 = 0; r8 < 8; ++r8) { m_run[r8] = -1e30f; l_run[r8] = 0.0f; }
#pragma unroll
  for (int q4 = 0; q4 < 4; ++q4)
#pragma unroll
    for (int r8 = 0; r8 < 8; ++r8) accO[q4][r8] = 0.0f;

  issue(0, 0);
  int buf = 0;
  for (int kb = 0; kb < n; kb += 32) {
    if (kb + 32 < n) {
      issue(kb + 32, buf ^ 1);
      asm volatile("s_wait_asynccnt 0x10" ::: "memory");  // current buffer's 16 loads done
    } else {
      asm volatile("s_wait_asynccnt 0x0" ::: "memory");
    }

    // ---- scores S = Qscaled @ K^T for 32 keys (two 16-key halves)
    v8f s[2];
#pragma unroll
    for (int nh2 = 0; nh2 < 2; ++nh2) {
      v8f acc;
#pragma unroll
      for (int r8 = 0; r8 < 8; ++r8) acc[r8] = 0.0f;
#pragma unroll
      for (int c = 0; c < 2; ++c) {
        v16h b = *reinterpret_cast<const v16h*>(&kv_lds[buf][nh2 * 2 + c][lane << 4]);
        acc = wmma16(aq[c], b, acc);
      }
      s[nh2] = acc;
    }

    // ---- online softmax (row r lives in vgpr r across one 16-lane half)
    v8f p0, p1;
#pragma unroll
    for (int r8 = 0; r8 < 8; ++r8) {
      float mv = fmaxf(s[0][r8], s[1][r8]);
      mv = fmaxf(mv, __shfl_xor(mv, 1, 32));
      mv = fmaxf(mv, __shfl_xor(mv, 2, 32));
      mv = fmaxf(mv, __shfl_xor(mv, 4, 32));
      mv = fmaxf(mv, __shfl_xor(mv, 8, 32));
      float mnew = fmaxf(m_run[r8], mv);
      float corr = __expf(m_run[r8] - mnew);
      float e0 = __expf(s[0][r8] - mnew);
      float e1 = __expf(s[1][r8] - mnew);
      p0[r8] = e0; p1[r8] = e1;
      float ts = e0 + e1;
      ts += __shfl_xor(ts, 1, 32);
      ts += __shfl_xor(ts, 2, 32);
      ts += __shfl_xor(ts, 4, 32);
      ts += __shfl_xor(ts, 8, 32);
      l_run[r8] = l_run[r8] * corr + ts;
      m_run[r8] = mnew;
#pragma unroll
      for (int q4 = 0; q4 < 4; ++q4) accO[q4][r8] *= corr;
    }

    // ---- transpose P (C-layout -> A-layout) through LDS
    __syncthreads();
#pragma unroll
    for (int r8 = 0; r8 < 8; ++r8) {
      int row = r8 + rbase;
      lds_p[row * 32 + bn]      = (_Float16)p0[r8];
      lds_p[row * 32 + 16 + bn] = (_Float16)p1[r8];
    }
    __syncthreads();
    v16h pa;
#pragma unroll
    for (int v = 0; v < 8; ++v) {
      int kk = (v << 1) + kh;
      pa[2 * v]     = lds_p[bn * 32 + kk];
      pa[2 * v + 1] = lds_p[bn * 32 + kk + 1];
    }

    // ---- O += P @ V  (K-dim = 32 keys; output 64 dims = 4 tiles of 16)
#pragma unroll
    for (int q4 = 0; q4 < 4; ++q4) {
      v16h bv = *reinterpret_cast<const v16h*>(&kv_lds[buf][4 + q4][lane << 4]);
      accO[q4] = wmma16(pa, bv, accO[q4]);
    }
    buf ^= 1;
  }

  // ---- finalize and store
#pragma unroll
  for (int q4 = 0; q4 < 4; ++q4) {
#pragma unroll
    for (int r8 = 0; r8 < 8; ++r8) {
      int row = qrow0 + r8 + rbase;
      out[(size_t)row * DFEAT + h * HDIM + q4 * 16 + bn] = accO[q4][r8] / l_run[r8];
    }
  }
}

// ---------------------------------------------------------------- driver
extern "C" void kernel_launch(void* const* d_in, const int* in_sizes, int n_in,
                              void* d_out, int out_size, void* d_ws, size_t ws_size,
                              hipStream_t stream) {
  const float* x         = (const float*)d_in[0];
  const int*   edges     = (const int*)d_in[1];
  const float* gin_w1    = (const float*)d_in[2];
  const float* gin_b1    = (const float*)d_in[3];
  const float* gin_w2    = (const float*)d_in[4];
  const float* gin_b2    = (const float*)d_in[5];
  const float* bn_g      = (const float*)d_in[6];
  const float* bn_b      = (const float*)d_in[7];
  const float* cheb_w    = (const float*)d_in[8];
  const float* cheb_b    = (const float*)d_in[9];
  const float* attn_in_w = (const float*)d_in[10];
  const float* attn_in_b = (const float*)d_in[11];
  const float* attn_out_w= (const float*)d_in[12];
  const float* attn_out_b= (const float*)d_in[13];
  const float* fuse_w    = (const float*)d_in[14];
  const float* fuse_b    = (const float*)d_in[15];
  const float* ln_g      = (const float*)d_in[16];
  const float* ln_b      = (const float*)d_in[17];

  const int N = in_sizes[0] / DIMD;
  const int E = in_sizes[1] / 2;
  const int* src = edges;
  const int* dst = edges + E;
  const size_t DD = (size_t)DIMD * DIMD;
  const size_t ND = (size_t)N * DIMD;

  float* ws   = (float*)d_ws;
  float* hA   = ws + 0 * ND;
  float* hB   = ws + 1 * ND;
  float* tA   = ws + 2 * ND;
  float* tB   = ws + 3 * ND;
  float* tC   = ws + 4 * ND;
  float* cbuf = ws + 5 * ND;                  // N x 512
  float* qkv  = ws + 7 * ND;                  // N x 1536
  float* obuf = ws + 13 * ND;                 // N x 512
  _Float16* aPack = (_Float16*)(ws + 15 * ND);   // up to M*K = N*512 halves
  _Float16* bPack = (_Float16*)(ws + 16 * ND);   // up to 512*1536 halves
  _Float16* kPack = (_Float16*)(ws + 17 * ND);   // N*512 halves
  _Float16* vPack = (_Float16*)(ws + 18 * ND);   // N*512 halves
  float* deg  = ws + 19 * ND;
  float* dis  = deg + N;
  float* colsum   = dis + N;
  float* colsumsq = colsum + DIMD;
  float* oproj = hA;   // N x 512, reuses hA+hB after GIN
  float* fused = tA;   // N x 256, reuses tA after Chebyshev

  auto zero = [&](float* p, size_t n) {
    zero_kernel<<<dim3((unsigned)((n + 255) / 256)), dim3(256), 0, stream>>>(p, (int)n);
  };
  auto gemm = [&](const float* A, const float* W, const float* bias, float* OUT,
                  int M, int Kd, int Nc, int ldA, int ldC,
                  int relu, int accum, float* cs, float* cs2) {
    int ta = (M / 16) * (Kd / 32) * 32;
    pack_a_kernel<<<dim3((ta + 255) / 256), dim3(256), 0, stream>>>(A, aPack, M, Kd, ldA);
    int tb = (Kd / 32) * (Nc / 16) * 32;
    pack_b_kernel<<<dim3((tb + 255) / 256), dim3(256), 0, stream>>>(W, bPack, Kd, Nc);
    int tiles = (M / 16) * (Nc / 64);
    wmma_gemm_pk<<<dim3((tiles + 7) / 8), dim3(256), 0, stream>>>(
        aPack, bPack, bias, OUT, M, Kd, Nc, ldC, relu, accum, cs, cs2);
  };

  const int st = E * (DIMD / 4);
  const int sblk = (st + 255) / 256;
  const int ndblk = (int)((ND + 255) / 256);

  // ================= GIN encoder (3 layers) =================
  const float* hcur = x;
  for (int i = 0; i < 3; ++i) {
    zero(colsum, DIMD);
    zero(colsumsq, DIMD);
    // tA = h; tA += segment_sum(h[src] -> dst)
    copy_kernel<<<dim3(ndblk), dim3(256), 0, stream>>>(hcur, tA, (int)ND);
    scatter_add_kernel<<<dim3(sblk), dim3(256), 0, stream>>>(hcur, src, dst, nullptr, tA, E, DIMD);
    // hidden = relu((h+agg) @ W1 + b1)
    gemm(tA, gin_w1 + i * DD, gin_b1 + i * DIMD, tB, N, DIMD, DIMD, DIMD, DIMD, 1, 0, nullptr, nullptr);
    // z = hidden @ W2 + b2, fused column stats for BN
    gemm(tB, gin_w2 + i * DD, gin_b2 + i * DIMD, tC, N, DIMD, DIMD, DIMD, DIMD, 0, 0, colsum, colsumsq);
    float* outp; int ldo;
    if (i == 0)      { outp = hA;   ldo = DIMD; }
    else if (i == 1) { outp = hB;   ldo = DIMD; }
    else             { outp = cbuf; ldo = DFEAT; }  // mp -> cbuf[:, :256]
    bn_apply_kernel<<<dim3((N * DIMD + 255) / 256), dim3(256), 0, stream>>>(
        tC, colsum, colsumsq, bn_g + i * DIMD, bn_b + i * DIMD, outp, N, DIMD, ldo, (i < 2) ? 1 : 0);
    hcur = (i == 0) ? hA : hB;
  }

  // ================= Chebyshev spectral conv =================
  zero(deg, N);
  deg_kernel<<<dim3((E + 255) / 256), dim3(256), 0, stream>>>(dst, deg, E);
  dis_kernel<<<dim3((N + 255) / 256), dim3(256), 0, stream>>>(deg, dis, N);
  float* spec = cbuf + DIMD;  // cbuf[:, 256:512], ldC = 512
  gemm(x, cheb_w, cheb_b, spec, N, DIMD, DIMD, DIMD, DFEAT, 0, 0, nullptr, nullptr);
  // t1 = L~ x
  zero(tA, ND);
  scatter_add_kernel<<<dim3(sblk), dim3(256), 0, stream>>>(x, src, dst, dis, tA, E, DIMD);
  gemm(tA, cheb_w + 1 * DD, nullptr, spec, N, DIMD, DIMD, DIMD, DFEAT, 0, 1, nullptr, nullptr);
  // k=2: t2 = 2 L~ t1 - t0
  zero(tB, ND);
  scatter_add_kernel<<<dim3(sblk), dim3(256), 0, stream>>>(tA, src, dst, dis, tB, E, DIMD);
  combine_kernel<<<dim3(ndblk), dim3(256), 0, stream>>>(tB, x, (int)ND);
  gemm(tB, cheb_w + 2 * DD, nullptr, spec, N, DIMD, DIMD, DIMD, DFEAT, 0, 1, nullptr, nullptr);
  // k=3
  zero(tC, ND);
  scatter_add_kernel<<<dim3(sblk), dim3(256), 0, stream>>>(tB, src, dst, dis, tC, E, DIMD);
  combine_kernel<<<dim3(ndblk), dim3(256), 0, stream>>>(tC, tA, (int)ND);
  gemm(tC, cheb_w + 3 * DD, nullptr, spec, N, DIMD, DIMD, DIMD, DFEAT, 0, 1, nullptr, nullptr);
  // k=4 (reuse tA)
  zero(tA, ND);
  scatter_add_kernel<<<dim3(sblk), dim3(256), 0, stream>>>(tC, src, dst, dis, tA, E, DIMD);
  combine_kernel<<<dim3(ndblk), dim3(256), 0, stream>>>(tA, tB, (int)ND);
  gemm(tA, cheb_w + 4 * DD, nullptr, spec, N, DIMD, DIMD, DIMD, DFEAT, 0, 1, nullptr, nullptr);

  // ================= Fusion: MHA over all nodes =================
  gemm(cbuf, attn_in_w, attn_in_b, qkv, N, DFEAT, 3 * DFEAT, DFEAT, 3 * DFEAT, 0, 0, nullptr, nullptr);
  {
    int tk = NHEAD * (N / 16) * 2 * 32;
    pack_k_kernel<<<dim3((tk + 255) / 256), dim3(256), 0, stream>>>(qkv, kPack, N);
    int tv = NHEAD * (N / 32) * 4 * 32;
    pack_v_kernel<<<dim3((tv + 255) / 256), dim3(256), 0, stream>>>(qkv, vPack, N);
  }
  flash_attn_kernel<<<dim3(NHEAD * (N / 16)), dim3(32), 0, stream>>>(qkv, kPack, vPack, obuf, N);
  gemm(obuf, attn_out_w, attn_out_b, oproj, N, DFEAT, DFEAT, DFEAT, DFEAT, 0, 0, nullptr, nullptr);
  gemm(oproj, fuse_w, fuse_b, fused, N, DFEAT, DIMD, DFEAT, DIMD, 0, 0, nullptr, nullptr);
  layernorm_kernel<<<dim3(N), dim3(32), 0, stream>>>(fused, ln_g, ln_b, (float*)d_out, DIMD);
}
